// SpatialReductionAttention_21277267984413
// MI455X (gfx1250) — matrix operand order repaired and checked by reading, not compiled, not run on hardware
//
#include <hip/hip_runtime.h>

// ---------------------------------------------------------------------------
// SpatialReductionAttention for MI455X (gfx1250), wave32 + WMMA f16.
// B=4, N=4096, C=256, HEADS=8, HD=32, SR=2, h=w=64, n_kv=1024.
// ---------------------------------------------------------------------------

typedef __attribute__((ext_vector_type(16))) _Float16 v16h;
typedef __attribute__((ext_vector_type(8)))  float    v8f;

union Frag {
    uint4 u[2];
    v16h  v;
};

// ---------------------------------------------------------------------------
// f32 -> f16 convert
// ---------------------------------------------------------------------------
__global__ __launch_bounds__(256) void cvt_f16_kernel(const float* __restrict__ src,
                                                      _Float16* __restrict__ dst, int n) {
    int i = blockIdx.x * 256 + threadIdx.x;
    if (i < n) dst[i] = (_Float16)src[i];
}

// ---------------------------------------------------------------------------
// im2col for the stride-2 / kernel-2 "spatial reduction" conv.
// P[b*1024 + oy*32 + ox][i*4 + ky*2 + kx] = x[b][(2oy+ky)*64 + 2ox+kx][i]
// (column order matches sr_w's OIHW flattening exactly)
// ---------------------------------------------------------------------------
__global__ __launch_bounds__(256) void patch_kernel(const float* __restrict__ x,
                                                    _Float16* __restrict__ P) {
    int i = blockIdx.x * 256 + threadIdx.x;   // < 4096*1024
    int col = i & 1023;
    int row = i >> 10;
    int b  = row >> 10, p = row & 1023;
    int oy = p >> 5,    ox = p & 31;
    int ci = col >> 2, ky = (col >> 1) & 1, kx = col & 1;
    int n = (2 * oy + ky) * 64 + 2 * ox + kx;
    P[i] = (_Float16)x[((size_t)b * 4096 + n) * 256 + ci];
}

// ---------------------------------------------------------------------------
// Generic WMMA GEMM:  out[M,N] = A[M,K] @ W[N,K]^T + bias
// wave tile 16x64 (4 accumulators), workgroup 8 waves -> 128x64 tile.
// MODE 0/3: f32 row-major.  MODE 1: f16 [B,H,N,HD].  MODE 2: f16 K/V split.
// ---------------------------------------------------------------------------
template <int MODE>
__global__ __launch_bounds__(256) void gemm_wmma_kernel(
        const _Float16* __restrict__ A, const _Float16* __restrict__ W,
        const float* __restrict__ bias, float* __restrict__ outF,
        _Float16* __restrict__ outH, _Float16* __restrict__ outH2,
        int M, int N, int K) {
    const int lane = threadIdx.x & 31;
    const int wave = threadIdx.x >> 5;
    const int half = lane >> 4;
    const int l16  = lane & 15;
    const int tileM = blockIdx.y * 128 + wave * 16;
    const int nBase = blockIdx.x * 64;

    v8f acc[4] = {};
    const _Float16* aPtr = A + (size_t)(tileM + l16) * K;

    for (int kc = 0; kc < K; kc += 32) {
        Frag a;                                     // A: 16x32, two 16B runs/lane
        a.u[0] = *(const uint4*)(aPtr + kc + half * 8);
        a.u[1] = *(const uint4*)(aPtr + kc + 16 + half * 8);
#pragma unroll
        for (int f = 0; f < 4; ++f) {
            const int n = nBase + f * 16 + l16;
            const _Float16* wPtr = W + (size_t)n * K + kc + half * 8;
            Frag b;                                 // B: 32x16, two 16B runs/lane
            b.u[0] = *(const uint4*)(wPtr);
            b.u[1] = *(const uint4*)(wPtr + 16);
            acc[f] = __builtin_amdgcn_wmma_f32_16x16x32_f16(
                false, a.v, false, b.v, (short)0, acc[f], false, false);
        }
    }

#pragma unroll
    for (int f = 0; f < 4; ++f) {
        const int c  = nBase + f * 16 + l16;
        const float bv = bias[c];
#pragma unroll
        for (int r = 0; r < 8; ++r) {
            const int m = tileM + r + 8 * half;     // C/D layout: row = r + 8*half
            const float v = acc[f][r] + bv;
            if (MODE == 0 || MODE == 3) {
                outF[(size_t)m * N + c] = v;
            } else if (MODE == 1) {                 // Q -> [B, H, 4096, 32] f16
                int b = m >> 12, n = m & 4095;
                int hh = c >> 5, d = c & 31;
                outH[((size_t)(b * 8 + hh) * 4096 + n) * 32 + d] = (_Float16)v;
            } else {                                // KV -> K/V [B, H, 1024, 32] f16
                int b = m >> 10, n = m & 1023;
                int s = c >> 8, c2 = c & 255;
                int hh = c2 >> 5, d = c2 & 31;
                _Float16* dst = s ? outH2 : outH;
                dst[((size_t)(b * 8 + hh) * 1024 + n) * 32 + d] = (_Float16)v;
            }
        }
    }
}

// ---------------------------------------------------------------------------
// LayerNorm over C=256; one wave per row, 8 elements per lane.
// ---------------------------------------------------------------------------
__global__ __launch_bounds__(256) void ln_kernel(const float* __restrict__ X,
                                                 const float* __restrict__ g,
                                                 const float* __restrict__ be,
                                                 _Float16* __restrict__ Y) {
    const int lane = threadIdx.x & 31;
    const int wave = threadIdx.x >> 5;
    const int row  = blockIdx.x * 8 + wave;
    const float* x = X + (size_t)row * 256;
    float v[8], s = 0.f;
#pragma unroll
    for (int e = 0; e < 8; ++e) { v[e] = x[lane * 8 + e]; s += v[e]; }
#pragma unroll
    for (int m = 1; m < 32; m <<= 1) s += __shfl_xor(s, m);
    const float mu = s * (1.0f / 256.0f);
    float var = 0.f;
#pragma unroll
    for (int e = 0; e < 8; ++e) { float d = v[e] - mu; var += d * d; }
#pragma unroll
    for (int m = 1; m < 32; m <<= 1) var += __shfl_xor(var, m);
    const float rs = rsqrtf(var * (1.0f / 256.0f) + 1e-5f);
    _Float16* y = Y + (size_t)row * 256;
#pragma unroll
    for (int e = 0; e < 8; ++e) {
        int c = lane * 8 + e;
        y[c] = (_Float16)((v[e] - mu) * rs * g[c] + be[c]);
    }
}

// ---------------------------------------------------------------------------
// Flash attention per (b,h): Q[4096,32] x K[1024,32]^T, online softmax, x V.
// Workgroup: 8 waves x 16 q-rows = 128 q rows. K/V chunked 256 rows in LDS.
// ---------------------------------------------------------------------------
__global__ __launch_bounds__(256) void attn_kernel(const _Float16* __restrict__ Q,
                                                   const _Float16* __restrict__ Kb,
                                                   const _Float16* __restrict__ Vb,
                                                   _Float16* __restrict__ Out,
                                                   float scale) {
    __shared__ _Float16 Kc[256][32];     // K rows (kpos, hd)        16 KB
    __shared__ _Float16 Vt[32][256];     // V transposed (hd, kpos)  16 KB
    __shared__ _Float16 Ps[8][16][32];   // per-wave P relayout       8 KB

    const int lane = threadIdx.x & 31;
    const int wave = threadIdx.x >> 5;
    const int half = lane >> 4;
    const int l16  = lane & 15;
    const int bh   = blockIdx.y;                    // b*8 + h
    const int qBase = blockIdx.x * 128 + wave * 16;

    // Q fragment (A layout), loaded once.
    const _Float16* qPtr = Q + ((size_t)bh * 4096 + qBase + l16) * 32;
    Frag qa;
    qa.u[0] = *(const uint4*)(qPtr + half * 8);
    qa.u[1] = *(const uint4*)(qPtr + 16 + half * 8);

    v8f o0 = {}, o1 = {};
    float mrow[8], lrow[8];
#pragma unroll
    for (int r = 0; r < 8; ++r) { mrow[r] = -1e30f; lrow[r] = 0.f; }

    const _Float16* kPtr = Kb + (size_t)bh * 1024 * 32;
    const _Float16* vPtr = Vb + (size_t)bh * 1024 * 32;

    for (int chunk = 0; chunk < 4; ++chunk) {
        const int kpos0 = chunk * 256;
        {   // cooperative load: thread t owns K/V row kpos0+t
            const int t = threadIdx.x;
            const uint4* src = (const uint4*)(kPtr + (size_t)(kpos0 + t) * 32);
            uint4* dst = (uint4*)(&Kc[t][0]);
            dst[0] = src[0]; dst[1] = src[1]; dst[2] = src[2]; dst[3] = src[3];
            const _Float16* vsrc = vPtr + (size_t)(kpos0 + t) * 32;
#pragma unroll
            for (int d = 0; d < 32; ++d) Vt[d][t] = vsrc[d];
        }
        __syncthreads();

        for (int kcL = 0; kcL < 256; kcL += 32) {
            // B fragments for S = Q K^T : B[hd][kpos]
            Frag bk0, bk1;
            {
                const _Float16* p0 = &Kc[kcL + l16][half * 8];
                bk0.u[0] = *(const uint4*)(p0);
                bk0.u[1] = *(const uint4*)(p0 + 16);
                const _Float16* p1 = &Kc[kcL + 16 + l16][half * 8];
                bk1.u[0] = *(const uint4*)(p1);
                bk1.u[1] = *(const uint4*)(p1 + 16);
            }
            v8f zero = {};
            v8f s0 = __builtin_amdgcn_wmma_f32_16x16x32_f16(
                false, qa.v, false, bk0.v, (short)0, zero, false, false);
            v8f s1 = __builtin_amdgcn_wmma_f32_16x16x32_f16(
                false, qa.v, false, bk1.v, (short)0, zero, false, false);

            // online softmax in C-layout; row stats reduced over 16-lane halves
#pragma unroll
            for (int r = 0; r < 8; ++r) {
                float a = s0[r] * scale;
                float b = s1[r] * scale;
                float t = fmaxf(a, b);
                t = fmaxf(t, __shfl_xor(t, 1));
                t = fmaxf(t, __shfl_xor(t, 2));
                t = fmaxf(t, __shfl_xor(t, 4));
                t = fmaxf(t, __shfl_xor(t, 8));
                const float mNew = fmaxf(mrow[r], t);
                const float fct = __expf(mrow[r] - mNew);
                const float p0 = __expf(a - mNew);
                const float p1 = __expf(b - mNew);
                float ps = p0 + p1;
                ps += __shfl_xor(ps, 1);
                ps += __shfl_xor(ps, 2);
                ps += __shfl_xor(ps, 4);
                ps += __shfl_xor(ps, 8);
                lrow[r] = lrow[r] * fct + ps;
                mrow[r] = mNew;
                o0[r] *= fct;
                o1[r] *= fct;
                const int row = r + 8 * half;
                Ps[wave][row][l16]      = (_Float16)p0;
                Ps[wave][row][16 + l16] = (_Float16)p1;
            }
            asm volatile("s_wait_dscnt 0" ::: "memory");  // per-wave LDS RAW

            Frag pa;                       // P as A fragment (16 x 32)
            {
                const _Float16* pp = &Ps[wave][l16][0];
                pa.u[0] = *(const uint4*)(pp + half * 8);
                pa.u[1] = *(const uint4*)(pp + 16 + half * 8);
            }
            Frag v0f, v1f;                 // V as B fragments: B[kpos][hd]
            {
                const _Float16* pv0 = &Vt[l16][kcL + half * 8];
                v0f.u[0] = *(const uint4*)(pv0);
                v0f.u[1] = *(const uint4*)(pv0 + 16);
                const _Float16* pv1 = &Vt[16 + l16][kcL + half * 8];
                v1f.u[0] = *(const uint4*)(pv1);
                v1f.u[1] = *(const uint4*)(pv1 + 16);
            }
            o0 = __builtin_amdgcn_wmma_f32_16x16x32_f16(
                false, pa.v, false, v0f.v, (short)0, o0, false, false);
            o1 = __builtin_amdgcn_wmma_f32_16x16x32_f16(
                false, pa.v, false, v1f.v, (short)0, o1, false, false);
        }
        __syncthreads();
    }

    // epilogue: O /= l ; write [B, N, C] with c = h*32 + d (f16 for proj GEMM)
    const int b = bh >> 3, hh = bh & 7;
#pragma unroll
    for (int r = 0; r < 8; ++r) {
        const float inv = 1.0f / lrow[r];
        const int qrow = qBase + r + 8 * half;
        const size_t base = ((size_t)b * 4096 + qrow) * 256 + hh * 32;
        Out[base + l16]      = (_Float16)(o0[r] * inv);
        Out[base + 16 + l16] = (_Float16)(o1[r] * inv);
    }
}

// ---------------------------------------------------------------------------
// Host-side launch
// ---------------------------------------------------------------------------
extern "C" void kernel_launch(void* const* d_in, const int* in_sizes, int n_in,
                              void* d_out, int out_size, void* d_ws, size_t ws_size,
                              hipStream_t stream) {
    const float* x      = (const float*)d_in[0];
    const float* q_w    = (const float*)d_in[1];
    const float* q_b    = (const float*)d_in[2];
    const float* kv_w   = (const float*)d_in[3];
    const float* kv_b   = (const float*)d_in[4];
    const float* sr_w   = (const float*)d_in[5];
    const float* sr_b   = (const float*)d_in[6];
    const float* ln_g   = (const float*)d_in[7];
    const float* ln_b   = (const float*)d_in[8];
    const float* proj_w = (const float*)d_in[9];
    const float* proj_b = (const float*)d_in[10];
    float* out = (float*)d_out;

    char* ws = (char*)d_ws;
    size_t off = 0;
    auto alloc = [&](size_t bytes) -> char* {
        char* p = ws + off;
        off += (bytes + 255) & ~(size_t)255;
        return p;
    };
    _Float16* xh    = (_Float16*)alloc((size_t)16384 * 256 * 2);  // x f16
    _Float16* qwh   = (_Float16*)alloc((size_t)256 * 256 * 2);
    _Float16* srwh  = (_Float16*)alloc((size_t)256 * 1024 * 2);
    _Float16* kvwh  = (_Float16*)alloc((size_t)512 * 256 * 2);
    _Float16* pwh   = (_Float16*)alloc((size_t)256 * 256 * 2);
    _Float16* qh    = (_Float16*)alloc((size_t)16384 * 256 * 2);  // [B,H,N,HD]
    _Float16* patch = (_Float16*)alloc((size_t)4096 * 1024 * 2);  // im2col
    float*    xsr   = (float*)alloc((size_t)4096 * 256 * 4);      // conv out f32
    _Float16* xln   = (_Float16*)alloc((size_t)4096 * 256 * 2);   // LN out f16
    _Float16* kh    = (_Float16*)alloc((size_t)32 * 1024 * 32 * 2);
    _Float16* vh    = (_Float16*)alloc((size_t)32 * 1024 * 32 * 2);
    _Float16* atth  = (_Float16*)alloc((size_t)16384 * 256 * 2);  // attn out f16

    // 1) f16 conversions
    cvt_f16_kernel<<<(4194304 + 255) / 256, 256, 0, stream>>>(x, xh, 4194304);
    cvt_f16_kernel<<<(65536 + 255) / 256, 256, 0, stream>>>(q_w, qwh, 65536);
    cvt_f16_kernel<<<(262144 + 255) / 256, 256, 0, stream>>>(sr_w, srwh, 262144);
    cvt_f16_kernel<<<(131072 + 255) / 256, 256, 0, stream>>>(kv_w, kvwh, 131072);
    cvt_f16_kernel<<<(65536 + 255) / 256, 256, 0, stream>>>(proj_w, pwh, 65536);

    // 2) im2col for SR conv
    patch_kernel<<<(4194304 + 255) / 256, 256, 0, stream>>>(x, patch);

    // 3) Q projection: [16384,256] @ [256,256]^T -> q f16 [B,H,N,HD]
    gemm_wmma_kernel<1><<<dim3(256 / 64, 16384 / 128), 256, 0, stream>>>(
        xh, qwh, q_b, nullptr, qh, nullptr, 16384, 256, 256);

    // 4) SR conv as GEMM: [4096,1024] @ [256,1024]^T -> f32
    gemm_wmma_kernel<0><<<dim3(256 / 64, 4096 / 128), 256, 0, stream>>>(
        patch, srwh, sr_b, xsr, nullptr, nullptr, 4096, 256, 1024);

    // 5) LayerNorm -> f16
    ln_kernel<<<4096 / 8, 256, 0, stream>>>(xsr, ln_g, ln_b, xln);

    // 6) KV projection: [4096,256] @ [512,256]^T -> K/V f16 [B,H,1024,HD]
    gemm_wmma_kernel<2><<<dim3(512 / 64, 4096 / 128), 256, 0, stream>>>(
        xln, kvwh, kv_b, nullptr, kh, vh, 4096, 512, 256);

    // 7) flash attention per (b,h)
    attn_kernel<<<dim3(4096 / 128, 32), 256, 0, stream>>>(
        qh, kh, vh, atth, 0.17677669529663688f /* 32^-0.5 */);

    // 8) final projection: [16384,256] @ [256,256]^T -> f32 d_out
    gemm_wmma_kernel<3><<<dim3(256 / 64, 16384 / 128), 256, 0, stream>>>(
        atth, pwh, proj_b, out, nullptr, nullptr, 16384, 256, 256);
}
